// BaselinePointerHead_10222022164634
// MI455X (gfx1250) — matrix operand order repaired and checked by reading, not compiled, not run on hardware
//
#include <hip/hip_runtime.h>
#include <hip/hip_bf16.h>
#include <math.h>

#define D_EMB   1024
#define DM      1024
#define SLEN    4096
#define MLEN    200
#define BATCH   16
#define ROLE_DIM 64
#define QIN_DIM (D_EMB + 2*ROLE_DIM)   // 1152
#define M_TILE  32
#define KTOT    1024

typedef __attribute__((ext_vector_type(16))) __bf16 v16bf;
typedef __attribute__((ext_vector_type(8)))  float  v8f;

struct __attribute__((aligned(16))) U4 { unsigned int x, y, z, w; };
struct __attribute__((aligned(16))) F4 { float x, y, z, w; };
union FragU { U4 u[2]; v16bf v; };

__device__ __forceinline__ unsigned short f32_to_bf16_rne(float f) {
  unsigned int u = __float_as_uint(f);
  unsigned int r = u + 0x7FFFu + ((u >> 16) & 1u);
  return (unsigned short)(r >> 16);
}

// ---------------------------------------------------------------- weights
__global__ void convert_weights(const float* __restrict__ wk,
                                const float* __restrict__ uw1,
                                unsigned short* __restrict__ wb) {
  int i = blockIdx.x * blockDim.x + threadIdx.x;       // 0 .. 2M-1
  if (i >= 2 * 1024 * 1024) return;
  int r = i >> 10, e = i & 1023;
  float v = (r < 1024) ? wk[(size_t)r * 1024 + e]
                       : uw1[(size_t)(r - 1024) * 1024 + e];
  wb[i] = f32_to_bf16_rne(v);
}

// ---------------------------------------------------------------- query
__global__ void build_query(const float* __restrict__ x_ctx,
                            const int* __restrict__ pos_suf,
                            const int* __restrict__ role_ctx,
                            const int* __restrict__ role_tgt,
                            const int* __restrict__ mask_ctx,
                            const float* __restrict__ hist_role_emb,
                            const float* __restrict__ tgt_role_emb,
                            const float* __restrict__ start_state,
                            const float* __restrict__ wq,
                            float* __restrict__ qrot) {
  int b = blockIdx.x, t = threadIdx.x;
  __shared__ float q_in[QIN_DIM];
  __shared__ float qtmp[DM];
  __shared__ int s_len;
  if (t == 0) s_len = 0;
  __syncthreads();
  int partial = 0;
  for (int m = t; m < MLEN; m += blockDim.x) partial += mask_ctx[b * MLEN + m];
  if (partial) atomicAdd(&s_len, partial);
  __syncthreads();
  int length = s_len;
  int last = length - 1; if (last < 0) last = 0;
  bool have = length > 0;
  int role_last = role_ctx[b * MLEN + last];
  int rt = role_tgt[b];
  for (int e = t; e < D_EMB; e += blockDim.x)
    q_in[e] = have ? x_ctx[((size_t)b * MLEN + last) * D_EMB + e] : start_state[e];
  if (t < ROLE_DIM) {
    q_in[D_EMB + t] = have ? hist_role_emb[role_last * ROLE_DIM + t]
                           : start_state[D_EMB + t];
    q_in[D_EMB + ROLE_DIM + t] = tgt_role_emb[rt * ROLE_DIM + t];
  }
  __syncthreads();
  for (int d = t; d < DM; d += blockDim.x) {
    float s = 0.f;
    const float* wr = wq + (size_t)d * QIN_DIM;
    for (int e = 0; e < QIN_DIM; ++e) s += wr[e] * q_in[e];
    qtmp[d] = s;
  }
  __syncthreads();
  float anchor = (float)pos_suf[(size_t)b * SLEN] - 1.0f;
  const float lg = -1.7988946e-2f;   // -2*ln(10000)/1024
  const float scale = 0.03125f;      // 1/sqrt(1024), folded into q
  for (int j = t; j < DM / 2; j += blockDim.x) {
    float invf = __expf(lg * (float)j);
    float c, s;
    sincosf(anchor * invf, &s, &c);
    float q1 = qtmp[j], q2 = qtmp[j + DM / 2];
    qrot[(size_t)b * DM + j]          = (q1 * c - q2 * s) * scale;
    qrot[(size_t)b * DM + j + DM / 2] = (q1 * s + q2 * c) * scale;
  }
}

// ---------------------------------------------------------------- fused GEMM
__global__ __launch_bounds__(256) void fused_main(
    const float* __restrict__ x_suf, const int* __restrict__ pos_suf,
    const int* __restrict__ mask_suf, const float* __restrict__ ln_g,
    const float* __restrict__ ln_b, const int* __restrict__ role_tgt,
    const float* __restrict__ u_b1, const float* __restrict__ u_w2,
    const float* __restrict__ u_b2,
    const unsigned short* __restrict__ wb, const float* __restrict__ qrot,
    float* __restrict__ logits)
{
  extern __shared__ unsigned short sA[];   // 32 rows x 1024 bf16 = 64 KB
  __shared__ float qsh[DM];
  __shared__ float red[M_TILE];
  __shared__ int   spos[M_TILE];
  __shared__ int   smask[M_TILE];

  const int t = threadIdx.x;
  const int lane = t & 31;
  const int wave = t >> 5;
  const int rowbase = blockIdx.x * M_TILE;
  const int b = rowbase >> 12;             // 4096 rows per batch
  const int role = role_tgt[b];

  for (int i = t; i < DM; i += 256) qsh[i] = qrot[(size_t)b * DM + i];
  if (t < M_TILE) red[t] = 0.f;

  // phase 1: layernorm rows -> bf16 in LDS (wave w handles rows 4w..4w+3)
  for (int j = 0; j < 4; ++j) {
    int i = wave * 4 + j;
    int r = rowbase + i;
    const float* xp = x_suf + (size_t)r * D_EMB + lane * 32;
    float vals[32];
    float s = 0.f, s2 = 0.f;
    #pragma unroll
    for (int c4 = 0; c4 < 8; ++c4) {
      F4 f = *reinterpret_cast<const F4*>(xp + c4 * 4);
      vals[c4*4+0]=f.x; vals[c4*4+1]=f.y; vals[c4*4+2]=f.z; vals[c4*4+3]=f.w;
    }
    #pragma unroll
    for (int k = 0; k < 32; ++k) { s += vals[k]; s2 += vals[k] * vals[k]; }
    for (int m = 16; m > 0; m >>= 1) {
      s  += __shfl_xor(s,  m, 32);
      s2 += __shfl_xor(s2, m, 32);
    }
    float mu   = s * (1.f / 1024.f);
    float var  = s2 * (1.f / 1024.f) - mu * mu;
    float rstd = rsqrtf(var + 1e-5f);
    int e0 = lane * 32;
    #pragma unroll
    for (int k = 0; k < 32; ++k) {
      int e = e0 + k;
      float xs = (vals[k] - mu) * rstd * ln_g[e] + ln_b[e];
      sA[i * D_EMB + e] = f32_to_bf16_rne(xs);
    }
    if (lane == 0) { spos[i] = pos_suf[r]; smask[i] = mask_suf[r]; }
  }
  __syncthreads();

  // phase 2: WMMA GEMM [32 x 1024] x [1024 x 2048] + fused epilogue.
  float partA[8], partB[8];
  #pragma unroll
  for (int v = 0; v < 8; ++v) { partA[v] = 0.f; partB[v] = 0.f; }

  const int klaneA = (lane >> 4) << 3;   // A frag: K 0-7/16-23 vs 8-15/24-31
  const int kbaseB = (lane >> 4) << 3;   // B frag: K 0-7/16-23 vs 8-15/24-31
  const int rowA0  = lane & 15;
  const float lg = -1.7988946e-2f;
  const float inv_s2 = 0.70710678f;

  // per-tile epilogue (inlined, constant-indexed accumulators)
  auto epilogue_tile = [&](v8f a0v, v8f a1v, int cg) {
    if (cg < 1024) {
      // k-part: fold RoPE(k) into dot with anchor-roped, pre-scaled q
      int f = cg & 511;
      float invf = __expf(lg * (float)f);
      float qa = qsh[f], qb = qsh[f + 512];
      bool lowhalf = cg < 512;
      #pragma unroll
      for (int v = 0; v < 8; ++v) {
        int r0 = (lane < 16) ? v : (8 + v);
        float c, s;
        __sincosf((float)spos[r0] * invf, &s, &c);
        float w0 = lowhalf ? (qa * c + qb * s) : (qb * c - qa * s);
        partA[v] += a0v[v] * w0;
        __sincosf((float)spos[r0 + 16] * invf, &s, &c);
        float w1 = lowhalf ? (qa * c + qb * s) : (qb * c - qa * s);
        partB[v] += a1v[v] * w1;
      }
    } else {
      // h-part: exact gelu then weight by u_w2[role]
      int ch = cg - 1024;
      float b1 = u_b1[ch];
      float w2 = u_w2[role * 1024 + ch];
      #pragma unroll
      for (int v = 0; v < 8; ++v) {
        float h0 = a0v[v] + b1;
        partA[v] += 0.5f * h0 * (1.f + erff(h0 * inv_s2)) * w2;
        float h1 = a1v[v] + b1;
        partB[v] += 0.5f * h1 * (1.f + erff(h1 * inv_s2)) * w2;
      }
    }
  };

  #pragma unroll 1
  for (int g = 0; g < 4; ++g) {
    v8f acc[4][2];
    const unsigned short* bp[4];
    int colg[4];
    #pragma unroll
    for (int j = 0; j < 4; ++j) {
      acc[j][0] = (v8f){}; acc[j][1] = (v8f){};
      int ct = wave + 8 * (4 * g + j);
      colg[j] = ct * 16 + (lane & 15);
      bp[j] = wb + (size_t)colg[j] * KTOT + kbaseB;
    }

    #pragma unroll 1
    for (int k0 = 0; k0 < KTOT; k0 += 32) {
      FragU a0, a1;
      int offA = rowA0 * D_EMB + k0 + klaneA;
      a0.u[0] = *reinterpret_cast<const U4*>(&sA[offA]);
      a0.u[1] = *reinterpret_cast<const U4*>(&sA[offA + 16]);
      a1.u[0] = *reinterpret_cast<const U4*>(&sA[offA + 16 * D_EMB]);
      a1.u[1] = *reinterpret_cast<const U4*>(&sA[offA + 16 * D_EMB + 16]);
      #pragma unroll
      for (int j = 0; j < 4; ++j) {
        FragU bf;
        bf.u[0] = *reinterpret_cast<const U4*>(bp[j] + k0);
        bf.u[1] = *reinterpret_cast<const U4*>(bp[j] + k0 + 16);
        acc[j][0] = __builtin_amdgcn_wmma_f32_16x16x32_bf16(
            false, a0.v, false, bf.v, (short)0, acc[j][0], false, false);
        acc[j][1] = __builtin_amdgcn_wmma_f32_16x16x32_bf16(
            false, a1.v, false, bf.v, (short)0, acc[j][1], false, false);
      }
    }

    // epilogue: fully unrolled so acc[] stays in registers (no scratch)
    #pragma unroll
    for (int j = 0; j < 4; ++j)
      epilogue_tile(acc[j][0], acc[j][1], colg[j]);
  }

  // reduce over the 16 columns held by each 16-lane half
  #pragma unroll
  for (int v = 0; v < 8; ++v)
    for (int m = 8; m > 0; m >>= 1) {
      partA[v] += __shfl_xor(partA[v], m, 32);
      partB[v] += __shfl_xor(partB[v], m, 32);
    }
  if ((lane & 15) == 0) {
    int rbase = (lane < 16) ? 0 : 8;
    #pragma unroll
    for (int v = 0; v < 8; ++v) {
      atomicAdd(&red[rbase + v],      partA[v]);   // rows 0..15
      atomicAdd(&red[rbase + 16 + v], partB[v]);   // rows 16..31
    }
  }
  __syncthreads();
  if (t < M_TILE) {
    float total = red[t] + u_b2[role];
    if (smask[t] == 0) total = -INFINITY;
    logits[(size_t)rowbase + t] = total;
  }
}

// ---------------------------------------------------------------- argmax
__global__ void argmax_kernel(const float* __restrict__ logits,
                              float* __restrict__ out_idx) {
  int b = blockIdx.x, t = threadIdx.x;
  __shared__ float sv[256];
  __shared__ int   si[256];
  float best = -INFINITY; int bi = 0x7fffffff;
  for (int s = t; s < SLEN; s += 256) {
    float v = logits[(size_t)b * SLEN + s];
    if (v > best || (v == best && s < bi)) { best = v; bi = s; }
  }
  sv[t] = best; si[t] = bi;
  __syncthreads();
  for (int off = 128; off > 0; off >>= 1) {
    if (t < off) {
      float v = sv[t + off]; int i2 = si[t + off];
      if (v > sv[t] || (v == sv[t] && i2 < si[t])) { sv[t] = v; si[t] = i2; }
    }
    __syncthreads();
  }
  if (t == 0) out_idx[b] = (float)(si[0] == 0x7fffffff ? 0 : si[0]);
}

// ---------------------------------------------------------------- launch
extern "C" void kernel_launch(void* const* d_in, const int* in_sizes, int n_in,
                              void* d_out, int out_size, void* d_ws, size_t ws_size,
                              hipStream_t stream) {
  (void)in_sizes; (void)n_in; (void)out_size; (void)ws_size;
  const float* x_ctx         = (const float*)d_in[0];
  const float* x_suf         = (const float*)d_in[1];
  const int*   pos_suf       = (const int*)d_in[3];
  const int*   role_ctx      = (const int*)d_in[4];
  const int*   role_tgt      = (const int*)d_in[5];
  const int*   mask_ctx      = (const int*)d_in[6];
  const int*   mask_suf      = (const int*)d_in[7];
  const float* ln_g          = (const float*)d_in[8];
  const float* ln_b          = (const float*)d_in[9];
  const float* hist_role_emb = (const float*)d_in[10];
  const float* tgt_role_emb  = (const float*)d_in[11];
  const float* start_state   = (const float*)d_in[12];
  const float* wq            = (const float*)d_in[13];
  const float* wk            = (const float*)d_in[14];
  const float* u_w1          = (const float*)d_in[15];
  const float* u_b1          = (const float*)d_in[16];
  const float* u_w2          = (const float*)d_in[17];
  const float* u_b2          = (const float*)d_in[18];

  unsigned short* wb = (unsigned short*)d_ws;                       // 4 MB bf16
  float* qrot = (float*)((char*)d_ws + (size_t)2*1024*1024*sizeof(unsigned short));
  float* logits  = (float*)d_out;
  float* out_idx = logits + (size_t)BATCH * SLEN;

  convert_weights<<<(2 * 1024 * 1024) / 256, 256, 0, stream>>>(wk, u_w1, wb);
  build_query<<<BATCH, 256, 0, stream>>>(x_ctx, pos_suf, role_ctx, role_tgt,
      mask_ctx, hist_role_emb, tgt_role_emb, start_state, wq, qrot);
  fused_main<<<(BATCH * SLEN) / M_TILE, 256,
               M_TILE * D_EMB * sizeof(unsigned short), stream>>>(
      x_suf, pos_suf, mask_suf, ln_g, ln_b, role_tgt,
      u_b1, u_w2, u_b2, wb, qrot, logits);
  argmax_kernel<<<BATCH, 256, 0, stream>>>(logits, out_idx);
}
